// GNNDecoder_16338055594319
// MI455X (gfx1250) — matrix operand order repaired and checked by reading, not compiled, not run on hardware
//
#include <hip/hip_runtime.h>
#include <hip/hip_bf16.h>

// ---------------------------------------------------------------------------
// CDNA5 (gfx1250) GNN decoder pipeline:
//   bf16 WMMA GEMMs with double-buffered GLOBAL_LOAD_ASYNC_TO_LDS_B128 tiles,
//   f32 atomic edge scatter, elementwise prep kernels for bf16 conversion.
// ---------------------------------------------------------------------------

typedef __bf16 bf16_t;
typedef bf16_t v16bf __attribute__((ext_vector_type(16)));
typedef float  v8f   __attribute__((ext_vector_type(8)));
typedef unsigned int v8u __attribute__((ext_vector_type(8)));
typedef int v4i __attribute__((ext_vector_type(4)));

#define AS1 __attribute__((address_space(1)))
#define AS3 __attribute__((address_space(3)))

#define TILE_M 128
#define TILE_N 64
#define TILE_K 32
#define LDS_STRIDE 40   // 32 + 8 halfword pad; 80 B/row keeps 16B-aligned chunks

// ---- f32 -> bf16 (RNE) ----------------------------------------------------
__device__ __forceinline__ unsigned short f32_to_bf16(float f) {
  unsigned u = __builtin_bit_cast(unsigned, f);
  u += 0x7FFFu + ((u >> 16) & 1u);
  return (unsigned short)(u >> 16);
}
__device__ __forceinline__ unsigned pack_bf16x2(float lo, float hi) {
  return (unsigned)f32_to_bf16(lo) | ((unsigned)f32_to_bf16(hi) << 16);
}

// ---- async memory -> LDS copy (16 bytes per lane) -------------------------
__device__ __forceinline__ void async_copy_b128(const void* gptr, void* lptr) {
#if __has_builtin(__builtin_amdgcn_global_load_async_to_lds_b128)
  __builtin_amdgcn_global_load_async_to_lds_b128(
      (AS1 v4i*)gptr, (AS3 v4i*)lptr, 0, 0);
#else
  unsigned loff = (unsigned)(unsigned long long)(AS3 void*)lptr;
  unsigned long long ga = (unsigned long long)gptr;
  asm volatile("global_load_async_to_lds_b128 %0, %1, off"
               :: "v"(loff), "v"(ga) : "memory");
#endif
}
__device__ __forceinline__ void wait_async_le3() {
#if __has_builtin(__builtin_amdgcn_s_wait_asynccnt)
  __builtin_amdgcn_s_wait_asynccnt(3);
#else
  asm volatile("s_wait_asynccnt 3" ::: "memory");
#endif
}
__device__ __forceinline__ void wait_async_0() {
#if __has_builtin(__builtin_amdgcn_s_wait_asynccnt)
  __builtin_amdgcn_s_wait_asynccnt(0);
#else
  asm volatile("s_wait_asynccnt 0" ::: "memory");
#endif
}

// Build one 8-VGPR (16 x bf16) fragment from two 16-byte LDS chunks.
__device__ __forceinline__ v16bf frag_from(const unsigned short* p0,
                                           const unsigned short* p1) {
  uint4 a = *(const uint4*)p0;
  uint4 b = *(const uint4*)p1;
  v8u u;
  u[0] = a.x; u[1] = a.y; u[2] = a.z; u[3] = a.w;
  u[4] = b.x; u[5] = b.y; u[6] = b.z; u[7] = b.w;
  return __builtin_bit_cast(v16bf, u);
}

// ---------------------------------------------------------------------------
// Tiled WMMA GEMM:  Out[M, Nout] = act(A[M, K] @ W[Nout, K]^T + bias)
// A and W are bf16. MODE_OUT: 0 = f32, 1 = f32 + bias, 2 = bf16 + bias + ReLU.
// Block = 256 threads (8 waves); block tile 128x64; each wave owns 32x32.
// K staged in steps of 32 through double-buffered LDS via async copies.
// ---------------------------------------------------------------------------
template <int K, int MODE_OUT>
__global__ __launch_bounds__(256) void gemm_wmma_kernel(
    const unsigned short* __restrict__ A, const unsigned short* __restrict__ Wb,
    const float* __restrict__ bias, void* __restrict__ Out, int M, int Nout) {
  __shared__ __align__(16) unsigned short ldsA[2][TILE_M * LDS_STRIDE];
  __shared__ __align__(16) unsigned short ldsB[2][TILE_N * LDS_STRIDE];

  const int tid   = threadIdx.x;
  const int lane  = tid & 31;
  const int wave  = tid >> 5;
  const int l16   = lane & 15;
  const int hi    = (lane >> 4) & 1;       // lanes 16..31 hold the high-K half
  const int mbase = (wave & 3) * 32;       // 4 waves along M
  const int nbase = (wave >> 2) * 32;      // 2 waves along N
  const int m0    = blockIdx.x * TILE_M;
  const int n0    = blockIdx.y * TILE_N;

  // ---- per-thread staging addresses (hoisted out of the K loop) ----
  // A tile 128x32 bf16 = 512 x 16B chunks, 2 per thread; B tile 64x32 = 256, 1.
  const int ca0 = tid, ca1 = tid + 256;
  const int rowA0 = ca0 >> 2, offA0 = (ca0 & 3) * 8;
  const int rowA1 = ca1 >> 2, offA1 = (ca1 & 3) * 8;
  const int rowB  = tid >> 2, offB  = (tid & 3) * 8;
  int gr0 = m0 + rowA0; gr0 = gr0 < M ? gr0 : (M - 1);  // clamp keeps EXEC full
  int gr1 = m0 + rowA1; gr1 = gr1 < M ? gr1 : (M - 1);
  const unsigned short* gA0 = A + (size_t)gr0 * K + offA0;
  const unsigned short* gA1 = A + (size_t)gr1 * K + offA1;
  const unsigned short* gB  = Wb + (size_t)(n0 + rowB) * K + offB;
  unsigned short* lA0[2] = {&ldsA[0][rowA0 * LDS_STRIDE + offA0],
                            &ldsA[1][rowA0 * LDS_STRIDE + offA0]};
  unsigned short* lA1[2] = {&ldsA[0][rowA1 * LDS_STRIDE + offA1],
                            &ldsA[1][rowA1 * LDS_STRIDE + offA1]};
  unsigned short* lB[2]  = {&ldsB[0][rowB * LDS_STRIDE + offB],
                            &ldsB[1][rowB * LDS_STRIDE + offB]};

  // ---- fragment LDS offsets (fixed per thread) ----
  const int rA0f = (mbase + 0  + l16) * LDS_STRIDE + hi * 8;
  const int rA1f = (mbase + 16 + l16) * LDS_STRIDE + hi * 8;
  const int rB0f = (nbase + 0  + l16) * LDS_STRIDE + hi * 8;
  const int rB1f = (nbase + 16 + l16) * LDS_STRIDE + hi * 8;

  v8f acc[2][2];
  acc[0][0] = 0.f; acc[0][1] = 0.f; acc[1][0] = 0.f; acc[1][1] = 0.f;

  // prologue: stage k-step 0 into buffer 0
  async_copy_b128(gA0, lA0[0]);
  async_copy_b128(gA1, lA1[0]);
  async_copy_b128(gB,  lB[0]);

  #pragma unroll 2
  for (int it = 0; it < K / TILE_K; ++it) {
    const int buf = it & 1;
    const bool has_next = (it + 1) < (K / TILE_K);
    if (has_next) {  // prefetch next k-step into the other buffer
      const int kn = (it + 1) * TILE_K;
      async_copy_b128(gA0 + kn, lA0[buf ^ 1]);
      async_copy_b128(gA1 + kn, lA1[buf ^ 1]);
      async_copy_b128(gB  + kn, lB[buf ^ 1]);
      wait_async_le3();  // async ops complete in order: current stage done
    } else {
      wait_async_0();
    }
    __syncthreads();

    const unsigned short* aB = ldsA[buf];
    const unsigned short* bB = ldsB[buf];
    // A 16x32 layout: lanes 0-15 -> M=lane, K {0..7,16..23}; lanes 16-31 -> +8
    v16bf fa0 = frag_from(&aB[rA0f], &aB[rA0f + 16]);
    v16bf fa1 = frag_from(&aB[rA1f], &aB[rA1f + 16]);
    // B 32x16 layout: lanes 0-15 -> N=lane, K {0..7,16..23}; lanes 16-31 -> +8
    v16bf fb0 = frag_from(&bB[rB0f], &bB[rB0f + 16]);
    v16bf fb1 = frag_from(&bB[rB1f], &bB[rB1f + 16]);

    acc[0][0] = __builtin_amdgcn_wmma_f32_16x16x32_bf16(
        false, fa0, false, fb0, (short)0, acc[0][0], false, false);
    acc[0][1] = __builtin_amdgcn_wmma_f32_16x16x32_bf16(
        false, fa0, false, fb1, (short)0, acc[0][1], false, false);
    acc[1][0] = __builtin_amdgcn_wmma_f32_16x16x32_bf16(
        false, fa1, false, fb0, (short)0, acc[1][0], false, false);
    acc[1][1] = __builtin_amdgcn_wmma_f32_16x16x32_bf16(
        false, fa1, false, fb1, (short)0, acc[1][1], false, false);
    __syncthreads();  // publish: buf may be overwritten next iteration
  }

  // ---- epilogue: C layout VGPR r -> M = (lane>=16 ? 8 : 0) + r, N = lane&15
  #pragma unroll
  for (int mi = 0; mi < 2; ++mi) {
    #pragma unroll
    for (int ni = 0; ni < 2; ++ni) {
      int colg = n0 + nbase + ni * 16 + l16;
      float bv = (MODE_OUT >= 1) ? bias[colg] : 0.f;
      #pragma unroll
      for (int r = 0; r < 8; ++r) {
        int rowg = m0 + mbase + mi * 16 + hi * 8 + r;
        if (rowg < M) {
          float v = acc[mi][ni][r] + bv;
          if (MODE_OUT == 2) {
            v = fmaxf(v, 0.f);
            ((unsigned short*)Out)[(size_t)rowg * Nout + colg] = f32_to_bf16(v);
          } else {
            ((float*)Out)[(size_t)rowg * Nout + colg] = v;
          }
        }
      }
    }
  }
}

// ---------------------------------------------------------------------------
// f32 -> bf16 conversion (optionally fused PReLU), 4 elements per thread.
// ---------------------------------------------------------------------------
template <bool PRELU>
__global__ __launch_bounds__(256) void convert_bf16_kernel(
    const float* __restrict__ in, unsigned short* __restrict__ out,
    const float* __restrict__ prelu_p, int total4) {
  int i = blockIdx.x * blockDim.x + threadIdx.x;
  if (i >= total4) return;
  float4 f = ((const float4*)in)[i];
  if (PRELU) {
    float a = *prelu_p;
    f.x = f.x >= 0.f ? f.x : a * f.x;
    f.y = f.y >= 0.f ? f.y : a * f.y;
    f.z = f.z >= 0.f ? f.z : a * f.z;
    f.w = f.w >= 0.f ? f.w : a * f.w;
  }
  uint2 p;
  p.x = pack_bf16x2(f.x, f.y);
  p.y = pack_bf16x2(f.z, f.w);
  ((uint2*)out)[i] = p;
}

// ---------------------------------------------------------------------------
// h1[mask[i]][:] = 0   (duplicates all write zero -> order-independent)
// ---------------------------------------------------------------------------
__global__ __launch_bounds__(256) void mask_zero_kernel(
    float* __restrict__ h1, const int* __restrict__ mask, int nMask) {
  int i = blockIdx.x * blockDim.x + threadIdx.x;
  if (i >= nMask * 256) return;
  h1[(size_t)mask[i >> 8] * 256 + (i & 255)] = 0.f;
}

// ---------------------------------------------------------------------------
// aggr[i][c] = h1[i][c] + E1[4][c] + E2[0][c]   (self-loop contribution)
// ---------------------------------------------------------------------------
__global__ __launch_bounds__(256) void aggr_init_kernel(
    const float* __restrict__ h1, const float* __restrict__ E1,
    const float* __restrict__ E2, float* __restrict__ aggr, int total) {
  int i = blockIdx.x * blockDim.x + threadIdx.x;
  if (i >= total) return;
  int c = i & 255;
  aggr[i] = h1[i] + E1[4 * 256 + c] + E2[c];
}

// ---------------------------------------------------------------------------
// One wave per edge: aggr[dst][:] += h1[src][:] + E1[a0][:] + E2[a1][:]
// 32 lanes x 2 float4 = 256 channels; f32 global atomics (resolve in L2).
// ---------------------------------------------------------------------------
__global__ __launch_bounds__(256) void edge_scatter_kernel(
    const float* __restrict__ h1, const int* __restrict__ ei,
    const int* __restrict__ ea, const float* __restrict__ E1,
    const float* __restrict__ E2, float* __restrict__ aggr, int E) {
  int e = blockIdx.x * (blockDim.x >> 5) + (threadIdx.x >> 5);
  int lane = threadIdx.x & 31;
  if (e >= E) return;
  int src = ei[e];
  int dst = ei[E + e];
  int a0  = ea[2 * e];
  int a1  = ea[2 * e + 1];
  const float4* hs = (const float4*)(h1 + (size_t)src * 256);
  const float4* g1 = (const float4*)(E1 + (size_t)a0 * 256);
  const float4* g2 = (const float4*)(E2 + (size_t)a1 * 256);
  float* out = aggr + (size_t)dst * 256;
  #pragma unroll
  for (int j = 0; j < 2; ++j) {
    int c = lane + j * 32;
    float4 v = hs[c];
    float4 w1 = g1[c];
    float4 w2 = g2[c];
    atomicAdd(out + 4 * c + 0, v.x + w1.x + w2.x);
    atomicAdd(out + 4 * c + 1, v.y + w1.y + w2.y);
    atomicAdd(out + 4 * c + 2, v.z + w1.z + w2.z);
    atomicAdd(out + 4 * c + 3, v.w + w1.w + w2.w);
  }
}

// ---------------------------------------------------------------------------
extern "C" void kernel_launch(void* const* d_in, const int* in_sizes, int n_in,
                              void* d_out, int out_size, void* d_ws,
                              size_t ws_size, hipStream_t stream) {
  const float* x       = (const float*)d_in[0];
  const int*   ei      = (const int*)d_in[1];
  const int*   ea      = (const int*)d_in[2];
  const int*   maskIdx = (const int*)d_in[3];
  const float* prelu_p = (const float*)d_in[4];
  const float* W_enc   = (const float*)d_in[5];
  const float* E1      = (const float*)d_in[6];
  const float* E2      = (const float*)d_in[7];
  const float* W1      = (const float*)d_in[8];
  const float* b1      = (const float*)d_in[9];
  const float* W2      = (const float*)d_in[10];
  const float* b2      = (const float*)d_in[11];

  const int HID = 256, MID = 512;
  const int N     = in_sizes[0] / HID;   // 20000
  const int E     = in_sizes[1] / 2;     // 320000
  const int NMASK = in_sizes[3];         // 2000

  // Workspace layout (xb aliases hmid: disjoint lifetimes).
  char* ws = (char*)d_ws;
  size_t off = 0;
  float* h1 = (float*)(ws + off);             off += (size_t)N * HID * 4;
  float* aggr = (float*)(ws + off);           off += (size_t)N * HID * 4;
  unsigned short* hmid = (unsigned short*)(ws + off); off += (size_t)N * MID * 2;
  unsigned short* xb = hmid;                  // [N,256] bf16, dead before hmid
  unsigned short* aggrb = (unsigned short*)(ws + off); off += (size_t)N * HID * 2;
  unsigned short* Wencb = (unsigned short*)(ws + off); off += (size_t)HID * HID * 2;
  unsigned short* W1b = (unsigned short*)(ws + off);   off += (size_t)MID * HID * 2;
  unsigned short* W2b = (unsigned short*)(ws + off);   off += (size_t)HID * MID * 2;

  dim3 blk(256);
  int mt = (N + TILE_M - 1) / TILE_M;

  // 0) precision prep: activations + weights -> bf16
  int t4x = N * HID / 4;
  convert_bf16_kernel<true><<<(t4x + 255) / 256, blk, 0, stream>>>(
      x, xb, prelu_p, t4x);
  int t4w0 = HID * HID / 4, t4w1 = MID * HID / 4;
  convert_bf16_kernel<false><<<(t4w0 + 255) / 256, blk, 0, stream>>>(
      W_enc, Wencb, nullptr, t4w0);
  convert_bf16_kernel<false><<<(t4w1 + 255) / 256, blk, 0, stream>>>(
      W1, W1b, nullptr, t4w1);
  convert_bf16_kernel<false><<<(t4w1 + 255) / 256, blk, 0, stream>>>(
      W2, W2b, nullptr, t4w1);

  // 1) h1 = prelu(x) @ W_enc^T  (f32 out, consumed by scatter)
  gemm_wmma_kernel<256, 0><<<dim3(mt, HID / TILE_N), blk, 0, stream>>>(
      xb, Wencb, nullptr, h1, N, HID);
  // 2) zero masked rows
  mask_zero_kernel<<<(NMASK * HID + 255) / 256, blk, 0, stream>>>(
      h1, maskIdx, NMASK);
  // 3) aggr = h1 + self-loop edge embedding
  aggr_init_kernel<<<(N * HID + 255) / 256, blk, 0, stream>>>(
      h1, E1, E2, aggr, N * HID);
  // 4) scatter edge messages (L2-resident atomics)
  edge_scatter_kernel<<<(E + 7) / 8, blk, 0, stream>>>(
      h1, ei, ea, E1, E2, aggr, E);
  // 4b) aggr -> bf16 for the next GEMM (halves its A-stream bytes)
  convert_bf16_kernel<false><<<(t4x + 255) / 256, blk, 0, stream>>>(
      aggr, aggrb, nullptr, t4x);
  // 5) h = relu(aggr @ W1^T + b1), stored bf16
  gemm_wmma_kernel<256, 2><<<dim3(mt, MID / TILE_N), blk, 0, stream>>>(
      aggrb, W1b, b1, hmid, N, MID);
  // 6) out = h @ W2^T + b2
  gemm_wmma_kernel<512, 1><<<dim3(mt, HID / TILE_N), blk, 0, stream>>>(
      hmid, W2b, b2, d_out, N, HID);
}
